// Sequence_2319282340086
// MI455X (gfx1250) — compile-verified, matrix operand order checked by reading, not run on hardware
//
#include <hip/hip_runtime.h>

// ---------------------------------------------------------------------------
// 2-layer LSTM (H=51), T=1024 + future autoregressive steps, B=512 (gfx1250).
// One workgroup (4 waves, 128 thr) per 16-row batch tile -> 32 independent
// persistent blocks. Gate GEMMs [16,64]x[64,208] on v_wmma_f32_16x16x32_f16;
// x_t and biases folded into the padded K dim (state col 51 = x, col 52 = 1,
// weight rows k=51/52 = Wi1 / bias) so the GEMM path is pure WMMA with
// inline-0 accumulator init. The 13 N-tiles split across 4 waves (scalar
// wave id keeps EXEC all-ones). Gating uses v_exp + v_rcp (no IEEE division),
// the output head is fused into layer-2 gating with a shfl_xor reduction,
// and x_{t+1} is prefetched at the top of each step.
// ---------------------------------------------------------------------------

typedef __attribute__((ext_vector_type(16))) _Float16 v16h;
typedef __attribute__((ext_vector_type(8)))  _Float16 v8h;
typedef __attribute__((ext_vector_type(8)))  float    v8f;

#define HSZ   51          // hidden size
#define GN    204         // 4*H gate columns
#define NTIL  13          // ceil(208/16)
#define TSEQ  1024        // sequence length (harness-fixed)
#define NW    4           // cooperating waves per workgroup

// fragment index: (g*13 + nt)*2 + kt ; g: 0=W_hh1(+Wi1+bias1), 1=W_ih2(+bias2), 2=W_hh2
#define NFRAG (3 * NTIL * 2)   // 78 fragments, 1KB each (f16)

// ---------------------------------------------------------------------------
// Pack weights -> f16 B-fragments with x-weight/bias folded into padded K.
// B[k][n] = W[n][k]. ISA 16-bit B layout: lane l: n = nt*16+(l&15),
// kb=(l>>4)*8; b[j]=B[kt*32+kb+j][n], b[8+j]=B[kt*32+16+kb+j][n].
// ---------------------------------------------------------------------------
__global__ void pack_weights_k(const float* __restrict__ Wh1,
                               const float* __restrict__ Wi1,
                               const float* __restrict__ bi1,
                               const float* __restrict__ bh1,
                               const float* __restrict__ Wi2,
                               const float* __restrict__ bi2,
                               const float* __restrict__ bh2,
                               const float* __restrict__ Wh2,
                               _Float16* __restrict__ wp)
{
    const int frag = blockIdx.x;    // 0..77
    const int lane = threadIdx.x;   // 0..31
    const int g  = frag / (NTIL * 2);
    const int r  = frag % (NTIL * 2);
    const int nt = r >> 1;
    const int kt = r & 1;

    const int nl = lane & 15;
    const int kb = (lane >> 4) * 8;
    const int n  = nt * 16 + nl;

    _Float16* dst = wp + ((size_t)frag * 32 + lane) * 16;
#pragma unroll
    for (int j = 0; j < 16; ++j) {
        const int k = kt * 32 + ((j < 8) ? (kb + j) : (16 + kb + (j - 8)));
        float v = 0.0f;
        if (n < GN) {
            if (g == 0) {                      // layer-1 recurrent + folded x/bias
                if      (k < HSZ)      v = Wh1[n * HSZ + k];
                else if (k == HSZ)     v = Wi1[n];            // x column
                else if (k == HSZ + 1) v = bi1[n] + bh1[n];   // bias column
            } else if (g == 1) {               // layer-2 input + folded bias
                if      (k < HSZ)      v = Wi2[n * HSZ + k];
                else if (k == HSZ + 1) v = bi2[n] + bh2[n];
            } else {                           // layer-2 recurrent
                if (k < HSZ)           v = Wh2[n * HSZ + k];
            }
        }
        dst[j] = (_Float16)v;
    }
}

// ---- device helpers -------------------------------------------------------

__device__ __forceinline__ v8f wmma_f16(v16h a, v16h b, v8f c) {
    return __builtin_amdgcn_wmma_f32_16x16x32_f16(
        false, a, false, b, (short)0, c, false, false);
}

// A-fragment (16x32 f16) for k-tile kt from row-major [16][64] f16 LDS state.
__device__ __forceinline__ v16h load_afrag(const _Float16* hs, int lane, int kt) {
    const int m  = lane & 15;
    const int kb = (lane >> 4) * 8;
    const _Float16* p = hs + m * 64 + kt * 32;
    v8h lo = *(const v8h*)(p + kb);
    v8h hi = *(const v8h*)(p + 16 + kb);
    v16h a;
#pragma unroll
    for (int j = 0; j < 8; ++j) { a[j] = lo[j]; a[8 + j] = hi[j]; }
    return a;
}

__device__ __forceinline__ v16h load_bfrag(const _Float16* __restrict__ wp,
                                           int fragIdx, int lane) {
    return *(const v16h*)(wp + ((size_t)fragIdx * 32 + lane) * 16);
}

// v_exp_f32 + v_rcp_f32 paths: no IEEE division sequences.
__device__ __forceinline__ float sigm(float x) {
    return __builtin_amdgcn_rcpf(1.0f + __expf(-x));
}
__device__ __forceinline__ float tanh_fast(float x) {
    return 2.0f * __builtin_amdgcn_rcpf(1.0f + __expf(-2.0f * x)) - 1.0f;
}

// ---------------------------------------------------------------------------
// Persistent LSTM kernel: 4 waves per 16 batch rows, full time loop.
// ---------------------------------------------------------------------------
__global__ __launch_bounds__(32 * NW)
void lstm_seq_k(const float* __restrict__ input,   // [B, TSEQ]
                const float* __restrict__ Wlin,    // [1,51]
                const float* __restrict__ blin,    // [1]
                const int*   __restrict__ futp,    // scalar future
                const _Float16* __restrict__ wp,   // packed fragments
                float* __restrict__ out)           // [B, TSEQ+future]
{
    __shared__ __align__(16) float    gates_s[16 * 208];
    __shared__ __align__(16) _Float16 h1s[16 * 64];  // [..50]=h1, [51]=x, [52]=1
    __shared__ __align__(16) _Float16 h2s[16 * 64];
    __shared__ __align__(16) float    c1s[16 * 64];
    __shared__ __align__(16) float    c2s[16 * 64];
    __shared__ float wlins[64];

    const int tid  = threadIdx.x;
    const int lane = tid & 31;
    const int wid  = __builtin_amdgcn_readfirstlane(tid >> 5); // SGPR wave id
    const int row0 = blockIdx.x * 16;
    const int F    = futp[0];
    const int Ttot = TSEQ + F;

    // ---- one-time init (includes staging x_0 into state col 51) ----
    for (int i = tid; i < 16 * 64; i += 32 * NW) {
        const int col = i & 63;
        h1s[i] = (col == HSZ + 1) ? (_Float16)1.0f : (_Float16)0.0f;
        h2s[i] = (_Float16)0.0f;
        c1s[i] = 0.0f; c2s[i] = 0.0f;
    }
    for (int i = tid; i < 64; i += 32 * NW) wlins[i] = (i < HSZ) ? Wlin[i] : 0.0f;
    const float bl = blin[0];
    if (tid < 16)
        h1s[tid * 64 + HSZ] = (_Float16)input[(size_t)(row0 + tid) * TSEQ];
    __builtin_prefetch(wp, 0, 1);   // global_prefetch_b8: warm packed weights
    __syncthreads();

    const int nl  = lane & 15;
    const int hi8 = (lane >> 4) * 8;     // C-matrix: VGPR r holds M = r + hi8
    const int m8  = tid >> 3;            // gating: fixed row per thread
    const int j0  = tid & 7;             // gating: column phase

    for (int t = 0; t < Ttot; ++t) {
        // ---- prefetch x_{t+1}; latency overlaps this whole step ----
        float xnext = 0.0f;
        if (tid < 16 && (t + 1) < TSEQ)
            xnext = input[(size_t)(row0 + tid) * TSEQ + (t + 1)];

        // ---- layer-1 gates: pure WMMA ([h1 | x | 1] @ [Wh1 | Wi1 | b1]) ----
        {
            const v16h a0 = load_afrag(h1s, lane, 0);
            const v16h a1 = load_afrag(h1s, lane, 1);
            for (int nt = wid; nt < NTIL; nt += NW) {   // scalar-strided tiles
                v8f acc = {};
                acc = wmma_f16(a0, load_bfrag(wp, (0 * NTIL + nt) * 2 + 0, lane), acc);
                acc = wmma_f16(a1, load_bfrag(wp, (0 * NTIL + nt) * 2 + 1, lane), acc);
                const int n = nt * 16 + nl;
#pragma unroll
                for (int r = 0; r < 8; ++r) gates_s[(r + hi8) * 208 + n] = acc[r];
            }
        }
        __syncthreads();

        // ---- layer-1 gating -> c1, h1(f16) ----
        {
            const float* gr = gates_s + m8 * 208;
            for (int j = j0; j < HSZ; j += 8) {
                const float si = sigm(gr[j]);
                const float sf = sigm(gr[51 + j]);
                const float tg = tanh_fast(gr[102 + j]);
                const float so = sigm(gr[153 + j]);
                const float c  = sf * c1s[m8 * 64 + j] + si * tg;
                c1s[m8 * 64 + j] = c;
                h1s[m8 * 64 + j] = (_Float16)(so * tanh_fast(c));
            }
        }
        __syncthreads();

        // ---- layer-2 gates: [h1|x|1]@[Wi2|0|b2] + h2@Wh2 ----
        {
            const v16h a10 = load_afrag(h1s, lane, 0);
            const v16h a11 = load_afrag(h1s, lane, 1);
            const v16h a20 = load_afrag(h2s, lane, 0);
            const v16h a21 = load_afrag(h2s, lane, 1);
            for (int nt = wid; nt < NTIL; nt += NW) {
                v8f acc = {};
                acc = wmma_f16(a10, load_bfrag(wp, (1 * NTIL + nt) * 2 + 0, lane), acc);
                acc = wmma_f16(a11, load_bfrag(wp, (1 * NTIL + nt) * 2 + 1, lane), acc);
                acc = wmma_f16(a20, load_bfrag(wp, (2 * NTIL + nt) * 2 + 0, lane), acc);
                acc = wmma_f16(a21, load_bfrag(wp, (2 * NTIL + nt) * 2 + 1, lane), acc);
                const int n = nt * 16 + nl;
#pragma unroll
                for (int r = 0; r < 8; ++r) gates_s[(r + hi8) * 208 + n] = acc[r];
            }
        }
        __syncthreads();

        // ---- layer-2 gating + fused output head + next-x commit ----
        {
            const float* gr = gates_s + m8 * 208;
            float part = 0.0f;
            for (int j = j0; j < HSZ; j += 8) {
                const float si = sigm(gr[j]);
                const float sf = sigm(gr[51 + j]);
                const float tg = tanh_fast(gr[102 + j]);
                const float so = sigm(gr[153 + j]);
                const float c  = sf * c2s[m8 * 64 + j] + si * tg;
                c2s[m8 * 64 + j] = c;
                const float h  = so * tanh_fast(c);
                h2s[m8 * 64 + j] = (_Float16)h;
                part += h * wlins[j];
            }
            // reduce across the 8 lanes of this row group (within one wave)
            part += __shfl_xor(part, 1);
            part += __shfl_xor(part, 2);
            part += __shfl_xor(part, 4);
            if (j0 == 0) {
                const float s = part + bl;
                out[(size_t)(row0 + m8) * Ttot + t] = s;
                if ((t + 1) >= TSEQ)                   // autoregressive feedback
                    h1s[m8 * 64 + HSZ] = (_Float16)s;
            }
            if (tid < 16 && (t + 1) < TSEQ)            // commit prefetched x
                h1s[tid * 64 + HSZ] = (_Float16)xnext;
        }
        __syncthreads();
    }
}

// ---------------------------------------------------------------------------
extern "C" void kernel_launch(void* const* d_in, const int* in_sizes, int n_in,
                              void* d_out, int out_size, void* d_ws, size_t ws_size,
                              hipStream_t stream)
{
    const float* input = (const float*)d_in[0];
    const float* Wi1   = (const float*)d_in[1];   // [204,1]
    const float* Wh1   = (const float*)d_in[2];   // [204,51]
    const float* bi1   = (const float*)d_in[3];
    const float* bh1   = (const float*)d_in[4];
    const float* Wi2   = (const float*)d_in[5];   // [204,51]
    const float* Wh2   = (const float*)d_in[6];   // [204,51]
    const float* bi2   = (const float*)d_in[7];
    const float* bh2   = (const float*)d_in[8];
    const float* Wlin  = (const float*)d_in[9];   // [1,51]
    const float* blin  = (const float*)d_in[10];  // [1]
    const int*   fut   = (const int*)d_in[11];    // scalar

    _Float16* wp  = (_Float16*)d_ws;              // 78 frags * 1KB = 79,872 B
    float*    out = (float*)d_out;

    const int B = in_sizes[0] / TSEQ;             // 512

    hipLaunchKernelGGL(pack_weights_k, dim3(NFRAG), dim3(32), 0, stream,
                       Wh1, Wi1, bi1, bh1, Wi2, bi2, bh2, Wh2, wp);

    hipLaunchKernelGGL(lstm_seq_k, dim3(B / 16), dim3(32 * NW), 0, stream,
                       input, Wlin, blin, fut, wp, out);
}
